// GraphMamba_9002251452653
// MI455X (gfx1250) — compile-verified, matrix operand order checked by reading, not run on hardware
//
#include <hip/hip_runtime.h>
#include <hip/hip_bf16.h>

// ---------------- problem constants (match reference) ----------------
#define NN      2048      // N nodes
#define CIN     16
#define DM      64
#define NL      4
#define DI      128
#define DS      16
#define DC      4
#define DTR     4
#define BV      16        // batch
#define EV      32768     // edges
#define TOK     (BV*NN)   // 32768 tokens
#define XZN     (2*DI)    // 256
#define XPN     36
#define XPP     48        // padded x-proj output cols (multiple of 16)
#define H1N     512
#define QN      32
#define KFLAT   (NN*DM)   // 131072
// scan chunking
#define CH      32        // chunks along L
#define CL      (NN/CH)   // 64 steps per chunk
// head1 split-K
#define KS      32        // K splits
#define KCH     (KFLAT/KS)// 4096 per split

typedef __attribute__((ext_vector_type(16))) __bf16 v16bf;
typedef __attribute__((ext_vector_type(8)))  __bf16 v8bf;
typedef __attribute__((ext_vector_type(8)))  float  v8f;

// ---- WMMA fragment loaders -------------------------------------------------
// 16-bit A/B frag, lane l: row = base + (l&15); K elements:
//   e in [0,8)  -> K = k0 + ((l&16)?8:0) + e
//   e in [8,16) -> K = k0 + 16 + ((l&16)?8:0) + (e-8)
// caller pre-offsets pointer by row*ld + ((l&16)?8:0); here we just add k0.
__device__ __forceinline__ v16bf frag_from_bf16(const __bf16* p) {
  v8bf lo = *(const v8bf*)p;
  v8bf hi = *(const v8bf*)(p + 16);
  v16bf f;
#pragma unroll
  for (int e = 0; e < 8; ++e) { f[e] = lo[e]; f[8 + e] = hi[e]; }
  return f;
}
__device__ __forceinline__ v16bf frag_from_f32(const float* p) {
  v8f lo = *(const v8f*)p;
  v8f hi = *(const v8f*)(p + 16);
  v16bf f;
#pragma unroll
  for (int e = 0; e < 8; ++e) { f[e] = (__bf16)lo[e]; f[8 + e] = (__bf16)hi[e]; }
  return f;
}

// ---------------- small utility kernels -------------------------------------
__global__ void k_zero_i32(int* p, int n) {
  int i = blockIdx.x * blockDim.x + threadIdx.x;
  if (i < n) p[i] = 0;
}
__global__ void k_cvt_bf16(const float* __restrict__ s, __bf16* __restrict__ d, int n) {
  int i = blockIdx.x * blockDim.x + threadIdx.x;
  if (i < n) d[i] = (__bf16)s[i];
}
// xp_w (NL,36,128) -> padded bf16 (NL,48,128), pad rows zero
__global__ void k_cvt_xp_pad(const float* __restrict__ s, __bf16* __restrict__ d) {
  int i = blockIdx.x * blockDim.x + threadIdx.x;       // over NL*48*128
  if (i >= NL * XPP * DI) return;
  int dd = i % DI; int e = (i / DI) % XPP; int l = i / (DI * XPP);
  float v = (e < XPN) ? s[(l * XPN + e) * DI + dd] : 0.0f;
  d[i] = (__bf16)v;
}

// ---------------- degree + stable rank sort ---------------------------------
__global__ void k_degree(const int* __restrict__ ei, int* __restrict__ deg) {
  int i = blockIdx.x * blockDim.x + threadIdx.x;
  if (i < EV) atomicAdd(&deg[ei[i]], 1);   // row = edge_index[0] = first E entries
}
// sort_idx = stable argsort of -deg ; unsort = inverse permutation
__global__ void k_rank_sort(const int* __restrict__ deg,
                            int* __restrict__ sort_idx, int* __restrict__ unsort_idx) {
  __shared__ int sdeg[NN];
  int t = threadIdx.x;
  for (int i = t; i < NN; i += blockDim.x) sdeg[i] = deg[i];
  __syncthreads();
  for (int i = t; i < NN; i += blockDim.x) {
    int di = sdeg[i], rank = 0;
    for (int j = 0; j < NN; ++j) {
      int dj = sdeg[j];
      rank += (dj > di) || (dj == di && j < i);
    }
    sort_idx[rank] = i;
    unsort_idx[i] = rank;
  }
}

// ---------------- shared LayerNorm helper ------------------------------------
__device__ __forceinline__ void ln_block(float v, const float* w, const float* b,
                                         int t, float* red, float* out) {
  red[t] = v; __syncthreads();
#pragma unroll
  for (int off = 32; off > 0; off >>= 1) { if (t < off) red[t] += red[t + off]; __syncthreads(); }
  float mean = red[0] / (float)DM; __syncthreads();
  float c = v - mean;
  red[t] = c * c; __syncthreads();
#pragma unroll
  for (int off = 32; off > 0; off >>= 1) { if (t < off) red[t] += red[t + off]; __syncthreads(); }
  float var = red[0] / (float)DM;
  *out = c * rsqrtf(var + 1e-5f) * w[t] + b[t];
}

// ---------------- embed (gather + 16->64) + LayerNorm -----------------------
__global__ void k_embed_ln(const float* __restrict__ x, const int* __restrict__ sidx,
                           const float* __restrict__ ew, const float* __restrict__ eb,
                           const float* __restrict__ lw, const float* __restrict__ lb,
                           float* __restrict__ h, __bf16* __restrict__ hbf) {
  int tok = blockIdx.x;            // b*NN + l  (sorted order)
  int t = threadIdx.x;             // 0..63
  int b = tok >> 11, l = tok & (NN - 1);
  __shared__ float xin[CIN];
  __shared__ float red[DM];
  int src = sidx[l];
  if (t < CIN) xin[t] = x[((size_t)b * NN + src) * CIN + t];
  __syncthreads();
  float v = eb[t];
#pragma unroll
  for (int c = 0; c < CIN; ++c) v += ew[t * CIN + c] * xin[c];
  float o;
  ln_block(v, lw, lb, t, red, &o);
  size_t idx = (size_t)tok * DM + t;
  h[idx] = o; hbf[idx] = (__bf16)o;
}

// ---------------- residual + LayerNorm --------------------------------------
__global__ void k_resid_ln(float* __restrict__ h, __bf16* __restrict__ hbf,
                           const float* __restrict__ yout,
                           const float* __restrict__ nw, const float* __restrict__ nb) {
  int tok = blockIdx.x; int t = threadIdx.x;
  __shared__ float red[DM];
  size_t idx = (size_t)tok * DM + t;
  float v = h[idx] + yout[idx];
  float o;
  ln_block(v, nw, nb, t, red, &o);
  h[idx] = o; hbf[idx] = (__bf16)o;
}

// ---------------- generic WMMA bf16 GEMM ------------------------------------
// C[M,Nn] = A[M,K] @ W^T  (W stored (Nn,K) row-major), f32 accumulate.
// one wave per 16x16 tile.
__global__ void k_gemm_bf16(const __bf16* __restrict__ A, const __bf16* __restrict__ W,
                            float* __restrict__ C, int K, int Nn, int ntiles) {
  int tile = blockIdx.x;
  int mtile = tile / ntiles;
  int ntile = tile - mtile * ntiles;
  int lane = threadIdx.x;
  int r16 = lane & 15;
  int khalf = (lane & 16) ? 8 : 0;
  const __bf16* arow = A + (size_t)(mtile * 16 + r16) * K + khalf;
  const __bf16* wrow = W + (size_t)(ntile * 16 + r16) * K + khalf;
  v8f acc = {};
  for (int k0 = 0; k0 < K; k0 += 32) {
    v16bf a = frag_from_bf16(arow + k0);
    v16bf b = frag_from_bf16(wrow + k0);
    acc = __builtin_amdgcn_wmma_f32_16x16x32_bf16(false, a, false, b, (short)0, acc,
                                                  false, false);
  }
  size_t cbase = (size_t)(mtile * 16 + khalf) * Nn + (size_t)ntile * 16 + r16;
#pragma unroll
  for (int r = 0; r < 8; ++r) C[cbase + (size_t)r * Nn] = acc[r];
}

// ---------------- depthwise causal conv (DC=4) + SiLU -----------------------
__global__ void k_conv_silu(const float* __restrict__ xz, const float* __restrict__ cw,
                            const float* __restrict__ cb,
                            float* __restrict__ xic, __bf16* __restrict__ xibf) {
  size_t i = (size_t)blockIdx.x * blockDim.x + threadIdx.x;  // TOK*DI
  if (i >= (size_t)TOK * DI) return;
  int d = (int)(i & (DI - 1));
  size_t tok = i >> 7;
  int b = (int)(tok >> 11), l = (int)(tok & (NN - 1));
  const float* w = cw + d * DC;
  float acc = cb[d];
#pragma unroll
  for (int t = 0; t < DC; ++t) {
    int ls = l - (DC - 1) + t;
    if (ls >= 0) acc += w[t] * xz[((size_t)b * NN + ls) * XZN + d];  // xi = xz[:, :DI]
  }
  float s = acc / (1.0f + __expf(-acc));   // SiLU
  xic[i] = s; xibf[i] = (__bf16)s;
}

// ---------------- dt projection (K=4) + softplus ----------------------------
__global__ void k_dt(const float* __restrict__ xdbl, const float* __restrict__ dtw,
                     const float* __restrict__ dtb, float* __restrict__ dt) {
  size_t i = (size_t)blockIdx.x * blockDim.x + threadIdx.x;  // TOK*DI
  if (i >= (size_t)TOK * DI) return;
  int d = (int)(i & (DI - 1));
  size_t tok = i >> 7;
  const float* xd = xdbl + tok * XPP;
  const float* w = dtw + d * DTR;
  float v = dtb[d];
#pragma unroll
  for (int r = 0; r < DTR; ++r) v += xd[r] * w[r];
  dt[i] = (v > 20.0f) ? v : log1pf(__expf(v));
}

// ---------------- chunked selective scan ------------------------------------
// Linear recurrence st_l = dA_l*st_{l-1} + dBx_l factored over CH chunks:
// pass1: per-chunk decay product P and zero-init partial state S
// pass2: 32-long prefix over chunks -> per-chunk initial states ST0
// pass3: replay each chunk from ST0, emit gated bf16 y
__global__ void k_scan_part(const float* __restrict__ dt, const float* __restrict__ xic,
                            const float* __restrict__ xdbl, const float* __restrict__ A_log,
                            float* __restrict__ P, float* __restrict__ S) {
  int blk = blockIdx.x;                 // b*CH + c
  int b = blk >> 5, c = blk & (CH - 1);
  int d = threadIdx.x;                  // 0..127
  const float* Al = A_log + d * DS;
  float a[DS], pr[DS], st[DS];
#pragma unroll
  for (int s = 0; s < DS; ++s) { a[s] = -__expf(Al[s]); pr[s] = 1.0f; st[s] = 0.0f; }
  for (int l = c * CL; l < (c + 1) * CL; ++l) {
    size_t tok = (size_t)b * NN + l;
    float dtv = dt[tok * DI + d];
    float dtx = dtv * xic[tok * DI + d];
    const float* xd = xdbl + tok * XPP;
#pragma unroll
    for (int s = 0; s < DS; ++s) {
      float dA = __expf(dtv * a[s]);
      st[s] = st[s] * dA + dtx * xd[DTR + s];
      pr[s] *= dA;
    }
  }
  size_t base = ((size_t)blk * DI + d) * DS;
#pragma unroll
  for (int s = 0; s < DS; ++s) { P[base + s] = pr[s]; S[base + s] = st[s]; }
}

__global__ void k_scan_prefix(const float* __restrict__ P, const float* __restrict__ S,
                              float* __restrict__ ST0) {
  int i = blockIdx.x * blockDim.x + threadIdx.x;   // BV*DI*DS = 32768
  if (i >= BV * DI * DS) return;
  int b = i / (DI * DS);
  int rem = i - b * (DI * DS);                     // d*DS + s
  float st = 0.0f;
  for (int c = 0; c < CH; ++c) {
    size_t base = (size_t)(b * CH + c) * (DI * DS) + rem;
    ST0[base] = st;
    st = S[base] + P[base] * st;
  }
}

__global__ void k_scan_fix(const float* __restrict__ dt, const float* __restrict__ xic,
                           const float* __restrict__ xdbl, const float* __restrict__ xz,
                           const float* __restrict__ A_log, const float* __restrict__ Dp,
                           const float* __restrict__ ST0, __bf16* __restrict__ ybf) {
  int blk = blockIdx.x;                 // b*CH + c
  int b = blk >> 5, c = blk & (CH - 1);
  int d = threadIdx.x;                  // 0..127
  const float* Al = A_log + d * DS;
  float a[DS], st[DS];
  size_t base = ((size_t)blk * DI + d) * DS;
#pragma unroll
  for (int s = 0; s < DS; ++s) { a[s] = -__expf(Al[s]); st[s] = ST0[base + s]; }
  float Dd = Dp[d];
  for (int l = c * CL; l < (c + 1) * CL; ++l) {
    size_t tok = (size_t)b * NN + l;
    float dtv = dt[tok * DI + d];
    float xv  = xic[tok * DI + d];
    float zv  = xz[tok * XZN + DI + d];
    const float* xd = xdbl + tok * XPP;
    float dtx = dtv * xv;
    float y = 0.0f;
#pragma unroll
    for (int s = 0; s < DS; ++s) {
      float dA = __expf(dtv * a[s]);
      st[s] = st[s] * dA + dtx * xd[DTR + s];       // Bm
      y += st[s] * xd[DTR + DS + s];                // Cm
    }
    y += xv * Dd;
    y *= zv / (1.0f + __expf(-zv));                 // * silu(z)
    ybf[tok * DI + d] = (__bf16)y;
  }
}

// ---------------- gather final h (unsort) as bf16 ---------------------------
__global__ void k_gather_final(const float* __restrict__ h, const int* __restrict__ uidx,
                               __bf16* __restrict__ hf) {
  size_t i = (size_t)blockIdx.x * blockDim.x + threadIdx.x;  // BV*KFLAT
  if (i >= (size_t)BV * KFLAT) return;
  int b = (int)(i / KFLAT);
  int j = (int)(i - (size_t)b * KFLAT);
  int n = j / DM, d = j - n * DM;
  hf[i] = (__bf16)h[((size_t)b * NN + uidx[n]) * DM + d];
}

// ---------------- head1 split-K: (16 x 131072) @ W1^T -----------------------
// grid (32 ntiles x KS ksplits): 1024 blocks / 8192 waves -> enough in-flight
// loads to stream head1_w (268MB) at HBM rate. Deterministic LDS reduction,
// partials to workspace, fixed-order final reduction (no float atomics).
__global__ void k_head1_part(const __bf16* __restrict__ Abf, const float* __restrict__ W1,
                             float* __restrict__ Hpart) {
  __shared__ float red[8 * 256];
  int wave = threadIdx.x >> 5;
  int lane = threadIdx.x & 31;
  int ntile = blockIdx.x;             // 0..31
  int ks    = blockIdx.y;             // 0..KS-1
  int r16 = lane & 15;
  int khalf = (lane & 16) ? 8 : 0;
  const __bf16* arow = Abf + (size_t)r16 * KFLAT + khalf;
  const float*  wrow = W1 + (size_t)(ntile * 16 + r16) * KFLAT + khalf;
  int kbeg = ks * KCH;
  int kend = kbeg + KCH;
  v8f acc = {};
  for (int k0 = kbeg + wave * 32; k0 < kend; k0 += 8 * 32) {
    if (k0 + 8 * 32 < kend)
      __builtin_prefetch(wrow + k0 + 8 * 32, 0, 1);   // global_prefetch_b8 on weight stream
    v16bf a = frag_from_bf16(arow + k0);
    v16bf b = frag_from_f32(wrow + k0);   // on-the-fly f32->bf16: keeps HBM traffic at 268MB
    acc = __builtin_amdgcn_wmma_f32_16x16x32_bf16(false, a, false, b, (short)0, acc,
                                                  false, false);
  }
#pragma unroll
  for (int r = 0; r < 8; ++r) red[wave * 256 + (r + khalf) * 16 + r16] = acc[r];
  __syncthreads();
  if (threadIdx.x < 256) {
    float s = 0.0f;
#pragma unroll
    for (int w = 0; w < 8; ++w) s += red[w * 256 + threadIdx.x];  // fixed order
    Hpart[((size_t)ks * 32 + ntile) * 256 + threadIdx.x] = s;
  }
}

__global__ void k_head1_reduce(const float* __restrict__ Hpart, const float* __restrict__ b1,
                               float* __restrict__ H1) {
  int i = blockIdx.x * blockDim.x + threadIdx.x;   // BV*H1N = 8192
  if (i >= BV * H1N) return;
  int m = i / H1N;
  int ng = i - m * H1N;
  int ntile = ng >> 4, n = ng & 15;
  float s = 0.0f;
  for (int ks = 0; ks < KS; ++ks)                  // fixed order: deterministic
    s += Hpart[((size_t)ks * 32 + ntile) * 256 + m * 16 + n];
  float v = s + b1[ng];
  H1[i] = v > 0.0f ? v : 0.0f;
}

// ---------------- head2: (16x512) @ W2^T + b2 -> out (16x32) ----------------
__global__ void k_head2(const float* __restrict__ H1, const float* __restrict__ W2,
                        const float* __restrict__ b2, float* __restrict__ out) {
  int i = blockIdx.x * blockDim.x + threadIdx.x;   // 16*32
  if (i >= BV * QN) return;
  int b = i >> 5, q = i & (QN - 1);
  const float* hr = H1 + (size_t)b * H1N;
  const float* wr = W2 + (size_t)q * H1N;
  float v = b2[q];
  for (int j = 0; j < H1N; ++j) v += hr[j] * wr[j];
  out[i] = v;
}

// =============================================================================
extern "C" void kernel_launch(void* const* d_in, const int* in_sizes, int n_in,
                              void* d_out, int out_size, void* d_ws, size_t ws_size,
                              hipStream_t stream) {
  const float* x       = (const float*)d_in[0];
  const int*   ei      = (const int*)  d_in[1];
  const float* emb_w   = (const float*)d_in[2];
  const float* emb_b   = (const float*)d_in[3];
  const float* ln_in_w = (const float*)d_in[4];
  const float* ln_in_b = (const float*)d_in[5];
  const float* in_w    = (const float*)d_in[6];
  const float* conv_w  = (const float*)d_in[7];
  const float* conv_b  = (const float*)d_in[8];
  const float* xp_w    = (const float*)d_in[9];
  const float* dt_w    = (const float*)d_in[10];
  const float* dt_b    = (const float*)d_in[11];
  const float* A_log   = (const float*)d_in[12];
  const float* Dp      = (const float*)d_in[13];
  const float* out_w   = (const float*)d_in[14];
  const float* norm_w  = (const float*)d_in[15];
  const float* norm_b  = (const float*)d_in[16];
  const float* h1w     = (const float*)d_in[17];
  const float* h1b     = (const float*)d_in[18];
  const float* h2w     = (const float*)d_in[19];
  const float* h2b     = (const float*)d_in[20];
  float* out = (float*)d_out;

  // ---- carve workspace ----
  char* p = (char*)d_ws;
  auto carve = [&](size_t bytes) { void* r = (void*)p; p += (bytes + 255) & ~(size_t)255; return r; };
  int*    deg     = (int*)   carve(NN * 4);
  int*    sidx    = (int*)   carve(NN * 4);
  int*    uidx    = (int*)   carve(NN * 4);
  float*  h       = (float*) carve((size_t)TOK * DM * 4);        // 8 MB
  __bf16* hbf     = (__bf16*)carve((size_t)TOK * DM * 2);        // 4 MB
  float*  xz      = (float*) carve((size_t)TOK * XZN * 4);       // 32 MB
  float*  xic     = (float*) carve((size_t)TOK * DI * 4);        // 16 MB
  __bf16* xibf    = (__bf16*)carve((size_t)TOK * DI * 2);        // 8 MB
  float*  xdbl    = (float*) carve((size_t)TOK * XPP * 4);       // 6 MB
  float*  dt      = (float*) carve((size_t)TOK * DI * 4);        // 16 MB
  __bf16* ybf     = (__bf16*)carve((size_t)TOK * DI * 2);        // 8 MB
  float*  yout    = (float*) carve((size_t)TOK * DM * 4);        // 8 MB
  float*  scanP   = (float*) carve((size_t)BV * CH * DI * DS * 4);   // 4 MB
  float*  scanS   = (float*) carve((size_t)BV * CH * DI * DS * 4);   // 4 MB
  float*  scanST0 = (float*) carve((size_t)BV * CH * DI * DS * 4);   // 4 MB
  __bf16* inw_bf  = (__bf16*)carve((size_t)NL * XZN * DM * 2);
  __bf16* xpw_bf  = (__bf16*)carve((size_t)NL * XPP * DI * 2);
  __bf16* outw_bf = (__bf16*)carve((size_t)NL * DM * DI * 2);
  __bf16* hfin_bf = (__bf16*)carve((size_t)BV * KFLAT * 2);      // 4 MB
  float*  Hpart   = (float*) carve((size_t)KS * 32 * 256 * 4);   // 1 MB
  float*  H1      = (float*) carve((size_t)BV * H1N * 4);

  // ---- weight conversions (once per call) ----
  {
    int n = NL * XZN * DM;
    k_cvt_bf16<<<(n + 255) / 256, 256, 0, stream>>>(in_w, inw_bf, n);
    n = NL * DM * DI;
    k_cvt_bf16<<<(n + 255) / 256, 256, 0, stream>>>(out_w, outw_bf, n);
    n = NL * XPP * DI;
    k_cvt_xp_pad<<<(n + 255) / 256, 256, 0, stream>>>(xp_w, xpw_bf);
  }

  // ---- degree + stable sort ----
  k_zero_i32<<<(NN + 255) / 256, 256, 0, stream>>>(deg, NN);
  k_degree<<<(EV + 255) / 256, 256, 0, stream>>>(ei, deg);
  k_rank_sort<<<1, 256, 0, stream>>>(deg, sidx, uidx);

  // ---- embed + input LN ----
  k_embed_ln<<<TOK, DM, 0, stream>>>(x, sidx, emb_w, emb_b, ln_in_w, ln_in_b, h, hbf);

  // ---- NL mamba layers ----
  const int MT = TOK / 16;   // 2048 M-tiles
  for (int i = 0; i < NL; ++i) {
    const float* Al = A_log + (size_t)i * DI * DS;
    // in-proj: (TOK,64) @ (256,64)^T -> xz (TOK,256)
    k_gemm_bf16<<<MT * (XZN / 16), 32, 0, stream>>>(hbf, inw_bf + (size_t)i * XZN * DM,
                                                    xz, DM, XZN, XZN / 16);
    // depthwise conv + SiLU
    {
      size_t n = (size_t)TOK * DI;
      k_conv_silu<<<(unsigned)((n + 255) / 256), 256, 0, stream>>>(
          xz, conv_w + (size_t)i * DI * DC, conv_b + (size_t)i * DI, xic, xibf);
    }
    // x-proj: (TOK,128) @ (48,128)^T -> xdbl (TOK,48)  [cols 36..47 zero pad]
    k_gemm_bf16<<<MT * (XPP / 16), 32, 0, stream>>>(xibf, xpw_bf + (size_t)i * XPP * DI,
                                                    xdbl, DI, XPP, XPP / 16);
    // dt projection + softplus
    {
      size_t n = (size_t)TOK * DI;
      k_dt<<<(unsigned)((n + 255) / 256), 256, 0, stream>>>(
          xdbl, dt_w + (size_t)i * DI * DTR, dt_b + (size_t)i * DI, dt);
    }
    // chunked selective scan (3 passes) + gate
    k_scan_part<<<BV * CH, DI, 0, stream>>>(dt, xic, xdbl, Al, scanP, scanS);
    k_scan_prefix<<<(BV * DI * DS + 255) / 256, 256, 0, stream>>>(scanP, scanS, scanST0);
    k_scan_fix<<<BV * CH, DI, 0, stream>>>(dt, xic, xdbl, xz, Al, Dp + (size_t)i * DI,
                                           scanST0, ybf);
    // out-proj: (TOK,128) @ (64,128)^T -> yout (TOK,64)
    k_gemm_bf16<<<MT * (DM / 16), 32, 0, stream>>>(ybf, outw_bf + (size_t)i * DM * DI,
                                                   yout, DI, DM, DM / 16);
    // residual + LN
    k_resid_ln<<<TOK, DM, 0, stream>>>(h, hbf, yout,
                                       norm_w + (size_t)i * DM, norm_b + (size_t)i * DM);
  }

  // ---- unsort + flatten to bf16 ----
  {
    size_t n = (size_t)BV * KFLAT;
    k_gather_final<<<(unsigned)((n + 255) / 256), 256, 0, stream>>>(h, uidx, hfin_bf);
  }
  // ---- head1 split-K WMMA + deterministic reduce ----
  k_head1_part<<<dim3(H1N / 16, KS), 256, 0, stream>>>(hfin_bf, h1w, Hpart);
  k_head1_reduce<<<(BV * H1N + 255) / 256, 256, 0, stream>>>(Hpart, h1b, H1);
  // ---- head2 ----
  k_head2<<<(BV * QN + 255) / 256, 256, 0, stream>>>(H1, h2w, h2b, out);
}
